// GINVirtualnode_63333587746873
// MI455X (gfx1250) — matrix operand order repaired and checked by reading, not compiled, not run on hardware
//
#include <hip/hip_runtime.h>
#include <hip/hip_bf16.h>

// ---------------------------------------------------------------------------
// GIN + virtual node on MI455X (gfx1250, wave32, WMMA bf16)
// ---------------------------------------------------------------------------

#define Nn   100000
#define Dd   128
#define Hh   256
#define Gg   256
#define Ll   5
#define Tt   128

typedef __attribute__((ext_vector_type(16))) __bf16 v16bf;
typedef __attribute__((ext_vector_type(8)))  float  v8f;

__device__ __forceinline__ unsigned short f2bf(float f) {
    unsigned int u = __float_as_uint(f);
    u += 0x7FFFu + ((u >> 16) & 1u);        // round-to-nearest-even
    return (unsigned short)(u >> 16);
}

// ---------------- elementwise / prep kernels -------------------------------

__global__ void k_f32_to_bf16(const float* __restrict__ in,
                              unsigned short* __restrict__ out, long n) {
    long i = (long)blockIdx.x * blockDim.x + threadIdx.x;
    if (i < n) out[i] = f2bf(in[i]);
}

// W [K][Nc] f32 row-major  ->  Wt [Nc][K] bf16 row-major
__global__ void k_transpose_bf16(const float* __restrict__ W,
                                 unsigned short* __restrict__ Wt,
                                 int K, int Nc) {
    long i = (long)blockIdx.x * blockDim.x + threadIdx.x;
    long n = (long)K * Nc;
    if (i < n) {
        int k = (int)(i / Nc), c = (int)(i % Nc);
        Wt[(long)c * K + k] = f2bf(W[i]);
    }
}

// BN fusion: scale = g*rsqrt(v+eps); shift = (bias - m)*scale + b
__global__ void k_scale_shift(const float* __restrict__ g, const float* __restrict__ b,
                              const float* __restrict__ m, const float* __restrict__ v,
                              const float* __restrict__ bias,
                              float* __restrict__ scale, float* __restrict__ shift, int n) {
    int i = blockIdx.x * blockDim.x + threadIdx.x;
    if (i < n) {
        float s = g[i] * rsqrtf(v[i] + 1e-5f);
        scale[i] = s;
        shift[i] = (bias[i] - m[i]) * s + b[i];
    }
}

__global__ void k_vn_init(const float* __restrict__ vn_init, float* __restrict__ vn, long n) {
    long i = (long)blockIdx.x * blockDim.x + threadIdx.x;
    if (i < n) vn[i] = vn_init[i & (Dd - 1)];
}

// hin = h + vn[batch[node]]
__global__ void k_add_vn(const float* __restrict__ h, const float* __restrict__ vn,
                         const int* __restrict__ batch, float* __restrict__ hin, long n) {
    long i = (long)blockIdx.x * blockDim.x + threadIdx.x;
    if (i < n) {
        int node = (int)(i >> 7), d = (int)(i & (Dd - 1));
        hin[i] = h[i] + vn[(long)batch[node] * Dd + d];
    }
}

// agg[dst[e]] += hin[src[e]]
// one wave per edge, 8 edges per 256-thread block; float4 loads + native fp32 atomics
__global__ void k_scatter(const float* __restrict__ hin, const int* __restrict__ src,
                          const int* __restrict__ dst, float* __restrict__ agg, int E) {
    int e = blockIdx.x * 8 + (threadIdx.x >> 5);
    if (e >= E) return;
    int lane = threadIdx.x & 31;
    int s = src[e], d = dst[e];
    const float4 v = *(const float4*)(hin + (long)s * Dd + lane * 4);
    float* ap = agg + (long)d * Dd + lane * 4;
    unsafeAtomicAdd(ap + 0, v.x);
    unsafeAtomicAdd(ap + 1, v.y);
    unsafeAtomicAdd(ap + 2, v.z);
    unsafeAtomicAdd(ap + 3, v.w);
}

// out_bf16 = bf16(a + b)
__global__ void k_add_cvt(const float* __restrict__ a, const float* __restrict__ b,
                          unsigned short* __restrict__ out, long n) {
    long i = (long)blockIdx.x * blockDim.x + threadIdx.x;
    if (i < n) out[i] = f2bf(a[i] + b[i]);
}

// pooled[batch[node]] += hin[node]   (also used for final graph sum)
// one lane handles 4 consecutive dims (float4 load, 4 native atomics)
__global__ void k_pool(const float* __restrict__ hin, const int* __restrict__ batch,
                       float* __restrict__ pooled, long nLanes /* Nn*32 */) {
    long i = (long)blockIdx.x * blockDim.x + threadIdx.x;
    if (i < nLanes) {
        int node = (int)(i >> 5), d4 = (int)(i & 31) * 4;
        const float4 v = *(const float4*)(hin + (long)node * Dd + d4);
        float* pp = pooled + (long)batch[node] * Dd + d4;
        unsafeAtomicAdd(pp + 0, v.x);
        unsafeAtomicAdd(pp + 1, v.y);
        unsafeAtomicAdd(pp + 2, v.z);
        unsafeAtomicAdd(pp + 3, v.w);
    }
}

__global__ void k_counts(const int* __restrict__ batch, float* __restrict__ counts, int n) {
    int i = blockIdx.x * blockDim.x + threadIdx.x;
    if (i < n) unsafeAtomicAdd(&counts[batch[i]], 1.0f);
}

// hg_bf16 = bf16(gsum / max(counts,1))
__global__ void k_mean_cvt(const float* __restrict__ gsum, const float* __restrict__ counts,
                           unsigned short* __restrict__ out, long n) {
    long i = (long)blockIdx.x * blockDim.x + threadIdx.x;
    if (i < n) {
        int g = (int)(i >> 7);
        out[i] = f2bf(gsum[i] / fmaxf(counts[g], 1.0f));
    }
}

// ---------------- WMMA GEMM -------------------------------------------------
// C[M][Nc] = epilogue( A[M][K]_bf16 @ Bt[Nc][K]_bf16 )
// one wave per 16x64 output strip; M % 16 == 0, Nc % 64 == 0, K % 32 == 0.
__global__ void k_gemm_wmma(const unsigned short* __restrict__ A,
                            const unsigned short* __restrict__ Bt,
                            int K, int Nc,
                            const float* __restrict__ scale,
                            const float* __restrict__ shift,
                            int relu,
                            float* __restrict__ Cf,
                            unsigned short* __restrict__ Cb) {
    const int lane = threadIdx.x;                 // 0..31
    const int row0 = blockIdx.x * 16;
    const int col0 = blockIdx.y * 64;
    const int aRow = row0 + (lane & 15);
    const int kA   = (lane >> 4) * 8;             // A frag K sub-offset
    const int kB   = (lane >> 4) * 8;             // B frag K sub-offset

    v8f acc[4] = {v8f{}, v8f{}, v8f{}, v8f{}};

    for (int k0 = 0; k0 < K; k0 += 32) {
        union { unsigned int u[8]; v16bf v; } a;
        const unsigned short* Ap = A + (long)aRow * K + k0;
#pragma unroll
        for (int j = 0; j < 4; ++j)
            a.u[j] = *(const unsigned int*)(Ap + kA + 2 * j);
#pragma unroll
        for (int j = 0; j < 4; ++j)
            a.u[4 + j] = *(const unsigned int*)(Ap + 16 + kA + 2 * j);

#pragma unroll
        for (int ct = 0; ct < 4; ++ct) {
            union { unsigned int u[8]; v16bf v; } b;
            const unsigned short* Bp =
                Bt + (long)(col0 + ct * 16 + (lane & 15)) * K + k0 + kB;
#pragma unroll
            for (int j = 0; j < 8; ++j)
                b.u[j] = *(const unsigned int*)(Bp + 2 * j + (j >= 4 ? 8 : 0));
            acc[ct] = __builtin_amdgcn_wmma_f32_16x16x32_bf16(
                false, a.v, false, b.v, (short)0, acc[ct], false, false);
        }
    }

    const int rBase = row0 + (lane >> 4) * 8;
    const int cLane = lane & 15;
#pragma unroll
    for (int ct = 0; ct < 4; ++ct) {
        int col  = col0 + ct * 16 + cLane;
        float s  = scale ? scale[col] : 1.0f;
        float sh = shift ? shift[col] : 0.0f;
#pragma unroll
        for (int j = 0; j < 8; ++j) {
            float v = acc[ct][j] * s + sh;
            if (relu) v = fmaxf(v, 0.0f);
            long o = (long)(rBase + j) * Nc + col;
            if (Cf) Cf[o] = v;
            if (Cb) Cb[o] = f2bf(v);
        }
    }
}

// ---------------- host side -------------------------------------------------

static inline long cdivl(long a, long b) { return (a + b - 1) / b; }

extern "C" void kernel_launch(void* const* d_in, const int* in_sizes, int n_in,
                              void* d_out, int out_size, void* d_ws, size_t ws_size,
                              hipStream_t stream) {
    (void)in_sizes; (void)n_in; (void)out_size; (void)ws_size;

    const float* x        = (const float*)d_in[0];
    const int*   ei       = (const int*)d_in[1];
    const int*   batch    = (const int*)d_in[2];
    const float* enc_W    = (const float*)d_in[3];
    const float* enc_b    = (const float*)d_in[4];
    const float* conv_W1  = (const float*)d_in[5];
    const float* conv_b1  = (const float*)d_in[6];
    const float* cbm_g    = (const float*)d_in[7];
    const float* cbm_b    = (const float*)d_in[8];
    const float* cbm_m    = (const float*)d_in[9];
    const float* cbm_v    = (const float*)d_in[10];
    const float* conv_W2  = (const float*)d_in[11];
    const float* conv_b2  = (const float*)d_in[12];
    const float* bn_g     = (const float*)d_in[13];
    const float* bn_b     = (const float*)d_in[14];
    const float* bn_m     = (const float*)d_in[15];
    const float* bn_v     = (const float*)d_in[16];
    const float* vn_W1    = (const float*)d_in[17];
    const float* vn_b1    = (const float*)d_in[18];
    const float* vbn1_g   = (const float*)d_in[19];
    const float* vbn1_b   = (const float*)d_in[20];
    const float* vbn1_m   = (const float*)d_in[21];
    const float* vbn1_v   = (const float*)d_in[22];
    const float* vn_W2    = (const float*)d_in[23];
    const float* vn_b2    = (const float*)d_in[24];
    const float* vbn2_g   = (const float*)d_in[25];
    const float* vbn2_b   = (const float*)d_in[26];
    const float* vbn2_m   = (const float*)d_in[27];
    const float* vbn2_v   = (const float*)d_in[28];
    const float* vn_init  = (const float*)d_in[29];
    const float* head_W   = (const float*)d_in[30];
    const float* head_b   = (const float*)d_in[31];

    const int E = in_sizes[1] / 2;
    const int* src = ei;
    const int* dst = ei + E;

    // ---- workspace bump allocator --------------------------------------
    char* p = (char*)d_ws;
    auto alloc = [&](size_t bytes) -> char* {
        char* r = p;
        p += (bytes + 255) & ~(size_t)255;
        return r;
    };
    float*          hA     = (float*)alloc((size_t)Nn * Dd * 4);
    float*          hB     = (float*)alloc((size_t)Nn * Dd * 4);
    float*          hin    = (float*)alloc((size_t)Nn * Dd * 4);
    float*          agg    = (float*)alloc((size_t)Nn * Dd * 4);
    unsigned short* tb     = (unsigned short*)alloc((size_t)Nn * Dd * 2);
    unsigned short* zb     = (unsigned short*)alloc((size_t)Nn * Hh * 2);
    float*          vn     = (float*)alloc((size_t)Gg * Dd * 4);
    float*          pooled = (float*)alloc((size_t)Gg * Dd * 4);
    unsigned short* pv     = (unsigned short*)alloc((size_t)Gg * Dd * 2);
    unsigned short* t2b    = (unsigned short*)alloc((size_t)Gg * Hh * 2);
    float*          counts = (float*)alloc((size_t)Gg * 4);
    unsigned short* encWt  = (unsigned short*)alloc((size_t)Dd * Dd * 2);
    unsigned short* W1t    = (unsigned short*)alloc((size_t)Ll * Dd * Hh * 2);
    unsigned short* W2t    = (unsigned short*)alloc((size_t)Ll * Hh * Dd * 2);
    unsigned short* vW1t   = (unsigned short*)alloc((size_t)(Ll - 1) * Dd * Hh * 2);
    unsigned short* vW2t   = (unsigned short*)alloc((size_t)(Ll - 1) * Hh * Dd * 2);
    unsigned short* headWt = (unsigned short*)alloc((size_t)Dd * Tt * 2);
    float* sc1 = (float*)alloc((size_t)Ll * Hh * 4);
    float* sh1 = (float*)alloc((size_t)Ll * Hh * 4);
    float* sc2 = (float*)alloc((size_t)Ll * Dd * 4);
    float* sh2 = (float*)alloc((size_t)Ll * Dd * 4);
    float* vs1 = (float*)alloc((size_t)(Ll - 1) * Hh * 4);
    float* vh1 = (float*)alloc((size_t)(Ll - 1) * Hh * 4);
    float* vs2 = (float*)alloc((size_t)(Ll - 1) * Dd * 4);
    float* vh2 = (float*)alloc((size_t)(Ll - 1) * Dd * 4);

    const long ND = (long)Nn * Dd;
    const long GD = (long)Gg * Dd;
    const long NL = (long)Nn * 32;        // pool lanes (4 dims each)
    dim3 b256(256);

    // ---- weight prep (every call; deterministic) -----------------------
    k_transpose_bf16<<<dim3(cdivl((long)Dd * Dd, 256)), b256, 0, stream>>>(enc_W, encWt, Dd, Dd);
    k_transpose_bf16<<<dim3(cdivl((long)Dd * Tt, 256)), b256, 0, stream>>>(head_W, headWt, Dd, Tt);
    for (int l = 0; l < Ll; ++l) {
        k_transpose_bf16<<<dim3(cdivl((long)Dd * Hh, 256)), b256, 0, stream>>>(
            conv_W1 + (long)l * Dd * Hh, W1t + (long)l * Dd * Hh, Dd, Hh);
        k_transpose_bf16<<<dim3(cdivl((long)Hh * Dd, 256)), b256, 0, stream>>>(
            conv_W2 + (long)l * Hh * Dd, W2t + (long)l * Hh * Dd, Hh, Dd);
        k_scale_shift<<<dim3(1), b256, 0, stream>>>(
            cbm_g + l * Hh, cbm_b + l * Hh, cbm_m + l * Hh, cbm_v + l * Hh,
            conv_b1 + l * Hh, sc1 + l * Hh, sh1 + l * Hh, Hh);
        k_scale_shift<<<dim3(1), b256, 0, stream>>>(
            bn_g + l * Dd, bn_b + l * Dd, bn_m + l * Dd, bn_v + l * Dd,
            conv_b2 + l * Dd, sc2 + l * Dd, sh2 + l * Dd, Dd);
    }
    for (int l = 0; l < Ll - 1; ++l) {
        k_transpose_bf16<<<dim3(cdivl((long)Dd * Hh, 256)), b256, 0, stream>>>(
            vn_W1 + (long)l * Dd * Hh, vW1t + (long)l * Dd * Hh, Dd, Hh);
        k_transpose_bf16<<<dim3(cdivl((long)Hh * Dd, 256)), b256, 0, stream>>>(
            vn_W2 + (long)l * Hh * Dd, vW2t + (long)l * Hh * Dd, Hh, Dd);
        k_scale_shift<<<dim3(1), b256, 0, stream>>>(
            vbn1_g + l * Hh, vbn1_b + l * Hh, vbn1_m + l * Hh, vbn1_v + l * Hh,
            vn_b1 + l * Hh, vs1 + l * Hh, vh1 + l * Hh, Hh);
        k_scale_shift<<<dim3(1), b256, 0, stream>>>(
            vbn2_g + l * Dd, vbn2_b + l * Dd, vbn2_m + l * Dd, vbn2_v + l * Dd,
            vn_b2 + l * Dd, vs2 + l * Dd, vh2 + l * Dd, Dd);
    }

    // ---- encoder: h = x @ enc_W + enc_b --------------------------------
    k_f32_to_bf16<<<dim3(cdivl(ND, 256)), b256, 0, stream>>>(x, tb, ND);
    k_gemm_wmma<<<dim3(Nn / 16, Dd / 64), dim3(32), 0, stream>>>(
        tb, encWt, Dd, Dd, nullptr, enc_b, 0, hA, nullptr);
    k_vn_init<<<dim3(cdivl(GD, 256)), b256, 0, stream>>>(vn_init, vn, GD);

    float* h = hA;
    float* hnew = hB;

    for (int l = 0; l < Ll; ++l) {
        // hin = h + vn[batch]
        k_add_vn<<<dim3(cdivl(ND, 256)), b256, 0, stream>>>(h, vn, batch, hin, ND);
        // agg = segment_sum(hin[src], dst)
        hipMemsetAsync(agg, 0, (size_t)ND * 4, stream);
        k_scatter<<<dim3(cdivl(E, 8)), b256, 0, stream>>>(hin, src, dst, agg, E);
        // tb = bf16(hin + agg)
        k_add_cvt<<<dim3(cdivl(ND, 256)), b256, 0, stream>>>(hin, agg, tb, ND);
        // z = relu(bn_mid(tb @ W1))  -> bf16
        k_gemm_wmma<<<dim3(Nn / 16, Hh / 64), dim3(32), 0, stream>>>(
            tb, W1t + (long)l * Dd * Hh, Dd, Hh, sc1 + l * Hh, sh1 + l * Hh, 1,
            nullptr, zb);
        // hnew = bn(z @ W2) (+relu if l < L-1) -> f32
        k_gemm_wmma<<<dim3(Nn / 16, Dd / 64), dim3(32), 0, stream>>>(
            zb, W2t + (long)l * Hh * Dd, Hh, Dd, sc2 + l * Dd, sh2 + l * Dd,
            (l < Ll - 1) ? 1 : 0, hnew, nullptr);

        if (l < Ll - 1) {
            // pooled = segment_sum(hin, batch); pv = bf16(pooled + vn)
            hipMemsetAsync(pooled, 0, (size_t)GD * 4, stream);
            k_pool<<<dim3(cdivl(NL, 256)), b256, 0, stream>>>(hin, batch, pooled, NL);
            k_add_cvt<<<dim3(cdivl(GD, 256)), b256, 0, stream>>>(pooled, vn, pv, GD);
            // t = relu(bn1(pv @ vnW1)) -> bf16
            k_gemm_wmma<<<dim3(Gg / 16, Hh / 64), dim3(32), 0, stream>>>(
                pv, vW1t + (long)l * Dd * Hh, Dd, Hh, vs1 + l * Hh, vh1 + l * Hh, 1,
                nullptr, t2b);
            // vn = relu(bn2(t @ vnW2)) -> f32
            k_gemm_wmma<<<dim3(Gg / 16, Dd / 64), dim3(32), 0, stream>>>(
                t2b, vW2t + (long)l * Hh * Dd, Hh, Dd, vs2 + l * Dd, vh2 + l * Dd, 1,
                vn, nullptr);
        }
        float* t = h; h = hnew; hnew = t;   // JK = 'last'
    }

    // ---- readout: mean pool + head -------------------------------------
    hipMemsetAsync(pooled, 0, (size_t)GD * 4, stream);
    hipMemsetAsync(counts, 0, (size_t)Gg * 4, stream);
    k_pool<<<dim3(cdivl(NL, 256)), b256, 0, stream>>>(h, batch, pooled, NL);
    k_counts<<<dim3(cdivl(Nn, 256)), b256, 0, stream>>>(batch, counts, Nn);
    k_mean_cvt<<<dim3(cdivl(GD, 256)), b256, 0, stream>>>(pooled, counts, pv, GD);
    k_gemm_wmma<<<dim3(Gg / 16, Tt / 64), dim3(32), 0, stream>>>(
        pv, headWt, Dd, Tt, nullptr, head_b, 0, (float*)d_out, nullptr);
}
